// DiscFace_AM_5205500363383
// MI455X (gfx1250) — matrix operand order repaired and checked by reading, not compiled, not run on hardware
//
#include <hip/hip_runtime.h>
#include <hip/hip_bf16.h>

#define BB 1024
#define DD 512
#define CC 100000
#define SCALE_ 64.0f
#define MARGIN_ 0.35f

typedef __attribute__((ext_vector_type(16))) _Float16 v16h;
typedef __attribute__((ext_vector_type(8)))  float    v8f;

union AFrag { v16h v; uint4 u[2]; };
union BFrag { v16h v; _Float16 h[16]; };
union CFrag { v8f  v; float f[8]; };

__device__ __forceinline__ float block_reduce_256(float v, float* red, int tid) {
    red[tid] = v; __syncthreads();
    #pragma unroll
    for (int s = 128; s > 0; s >>= 1) {
        if (tid < s) red[tid] += red[tid + s];
        __syncthreads();
    }
    float r = red[0]; __syncthreads();
    return r;
}

// One block (256 threads) per batch row:
//  - L2-normalize x row -> write f16 copy for WMMA
//  - gather w[target], b[target]: disc-loss term + exact fp32 target logit
__global__ __launch_bounds__(256)
void row_prep_kernel(const float* __restrict__ x, const int* __restrict__ target,
                     const float* __restrict__ wraw, const float* __restrict__ bvec,
                     _Float16* __restrict__ xh, float* __restrict__ straw,
                     float* __restrict__ disc_sum)
{
    __shared__ float red[256];
    const int m = blockIdx.x;
    const int tid = threadIdx.x;

    const float* xr = x + (size_t)m * DD;
    float x0 = xr[tid], x1 = xr[tid + 256];

    float xsq = block_reduce_256(x0 * x0 + x1 * x1, red, tid);
    float invx = 1.0f / fmaxf(sqrtf(xsq), 1e-12f);
    float xn0 = x0 * invx, xn1 = x1 * invx;
    xh[(size_t)m * DD + tid]       = (_Float16)xn0;
    xh[(size_t)m * DD + tid + 256] = (_Float16)xn1;

    const int t = target[m];
    const float* wr = wraw + (size_t)t * DD;
    const float* br = bvec + (size_t)t * DD;
    float w0 = wr[tid], w1 = wr[tid + 256];
    float b0 = br[tid], b1 = br[tid + 256];

    float wsq  = block_reduce_256(w0 * w0 + w1 * w1, red, tid);
    float btsq = block_reduce_256(b0 * b0 + b1 * b1, red, tid);
    float dot  = block_reduce_256(xn0 * w0 + xn1 * w1, red, tid);

    float invw = rsqrtf(fmaxf(wsq, 1e-24f));
    float btn  = sqrtf(btsq);
    float btc  = fminf(btn, 0.05f);                 // clip(btnorm, 0, 0.05)
    float ibu  = btc / fmaxf(btn, 1e-12f);          // (bt_unit * btnorm_c) scale
    float r0 = (xn0 - w0 * invw) - b0 * ibu;
    float r1 = (xn1 - w1 * invw) - b1 * ibu;
    float rsq = block_reduce_256(r0 * r0 + r1 * r1, red, tid);

    if (tid == 0) {
        atomicAdd(disc_sum, sqrtf(rsq));
        straw[m] = SCALE_ * dot * invw;             // unmargined target logit
    }
}

// One WG (16 waves / 512 threads) per 64-class tile; all 1024 rows.
// Per wave: 4 row-tiles x 4 class-tiles of 16x16 WMMA-f16 accumulators.
// Class norms accumulated from the same fp32 stream that feeds the B fragments.
// Out-of-range classes (last block only) are index-CLAMPED, not branched:
// the hot loop stays EXEC-uniform and branch-free; garbage contributions are
// masked in the epilogue.
__global__ __launch_bounds__(512)
void gemm_lse_kernel(const _Float16* __restrict__ xh,
                     const float* __restrict__ wraw,
                     float* __restrict__ sumexp)
{
    const int wave  = threadIdx.x >> 5;
    const int lane  = threadIdx.x & 31;
    const int ml    = lane & 15;       // A: M row / B,C,D: N column
    const int khi   = lane >> 4;       // K half selector
    const int cbase = blockIdx.x * 64;
    const int rowbase = wave * 64;

    CFrag acc[4][4] = {};
    float sq[4] = {0.f, 0.f, 0.f, 0.f};
    bool  vld[4];
    const float* wp0[4];
    #pragma unroll
    for (int ct = 0; ct < 4; ++ct) {
        int c   = cbase + ct * 16 + ml;
        vld[ct] = c < CC;
        int cc  = vld[ct] ? c : (CC - 1);           // clamp: always-valid loads
        wp0[ct] = wraw + (size_t)cc * DD + 16 * khi;
    }

    for (int k0 = 0; k0 < DD; k0 += 32) {
        // A fragments: x_n f16, 16x32 tile; two contiguous 16B chunks per lane
        AFrag a[4];
        #pragma unroll
        for (int rt = 0; rt < 4; ++rt) {
            const _Float16* ap =
                xh + (size_t)(rowbase + rt * 16 + ml) * DD + k0 + 16 * khi;
            a[rt].u[0] = *(const uint4*)(ap);
            a[rt].u[1] = *(const uint4*)(ap + 8);
        }
        #pragma unroll
        for (int ct = 0; ct < 4; ++ct) {
            // B fragment: 16 contiguous K-halves of class row, from fp32
            const float4* wp = (const float4*)(wp0[ct] + k0);
            float4 q0 = wp[0], q1 = wp[1], q2 = wp[2], q3 = wp[3];
            float f[16] = { q0.x, q0.y, q0.z, q0.w,  q1.x, q1.y, q1.z, q1.w,
                            q2.x, q2.y, q2.z, q2.w,  q3.x, q3.y, q3.z, q3.w };
            BFrag bf;
            #pragma unroll
            for (int j = 0; j < 16; ++j) {
                bf.h[j] = (_Float16)f[j];
                sq[ct] += f[j] * f[j];             // class norm, free side-effect
            }
            #pragma unroll
            for (int rt = 0; rt < 4; ++rt) {
                acc[rt][ct].v = __builtin_amdgcn_wmma_f32_16x16x32_f16(
                    false, a[rt].v, false, bf.v, (short)0, acc[rt][ct].v,
                    false, false);
            }
        }
    }

    // finish class norms: lane n holds K halves for khi; combine with lane n+16
    float invn[4];
    #pragma unroll
    for (int ct = 0; ct < 4; ++ct) {
        float tot = sq[ct] + __shfl_xor(sq[ct], 16, 32);
        invn[ct] = rsqrtf(fmaxf(tot, 1e-24f));
    }

    // Epilogue: score = 64*dot/||w||; sumexp partial with fixed max 64.
    // D layout: lane = n + 16*hi holds M = r + 8*hi in f[r].
    #pragma unroll
    for (int rt = 0; rt < 4; ++rt) {
        #pragma unroll
        for (int r = 0; r < 8; ++r) {
            float etot = 0.f;
            #pragma unroll
            for (int ct = 0; ct < 4; ++ct) {
                float s = SCALE_ * acc[rt][ct].f[r] * invn[ct];
                etot += vld[ct] ? __expf(s - SCALE_) : 0.f;
            }
            #pragma unroll
            for (int off = 1; off < 16; off <<= 1)
                etot += __shfl_xor(etot, off, 32);
            if (ml == 0)
                atomicAdd(&sumexp[rowbase + rt * 16 + r + 8 * khi], etot);
        }
    }
}

// Single block: margin fix-up, NLL per row, focal CE, combine with disc loss.
__global__ __launch_bounds__(1024)
void finalize_kernel(const float* __restrict__ straw,
                     const float* __restrict__ sumexp,
                     const float* __restrict__ disc_sum,
                     float* __restrict__ out)
{
    __shared__ float red[1024];
    const int tid = threadIdx.x;
    float st  = straw[tid];
    float stm = st - SCALE_ * MARGIN_;
    float se  = sumexp[tid] + expf(stm - SCALE_) - expf(st - SCALE_);
    float nll = SCALE_ + logf(fmaxf(se, 1e-30f)) - stm;

    red[tid] = nll; __syncthreads();
    #pragma unroll
    for (int s = 512; s > 0; s >>= 1) {
        if (tid < s) red[tid] += red[tid + s];
        __syncthreads();
    }
    if (tid == 0) {
        float logp = red[0] / (float)BB;
        float p = expf(-logp);
        float loss_score = (1.f - p) * (1.f - p) * logp;
        out[0] = 0.4f * (disc_sum[0] / (float)BB) + loss_score;
    }
}

extern "C" void kernel_launch(void* const* d_in, const int* in_sizes, int n_in,
                              void* d_out, int out_size, void* d_ws, size_t ws_size,
                              hipStream_t stream) {
    const float* x        = (const float*)d_in[0];
    const int*   target   = (const int*)  d_in[1];
    const float* id_agent = (const float*)d_in[2];
    const float* bvec     = (const float*)d_in[3];

    // workspace layout (16B-aligned f16 region)
    float*    sumexp = (float*)d_ws;                 // [1024]
    float*    disc   = sumexp + 1024;                // [1]   (padded)
    float*    straw  = sumexp + 2048;                // [1024] (padded)
    _Float16* xh     = (_Float16*)(sumexp + 4096);   // [1024*512] f16

    hipMemsetAsync(d_ws, 0, 2048 * sizeof(float), stream);

    row_prep_kernel<<<BB, 256, 0, stream>>>(x, target, id_agent, bvec,
                                            xh, straw, disc);
    gemm_lse_kernel<<<(CC + 63) / 64, 512, 0, stream>>>(xh, id_agent, sumexp);
    finalize_kernel<<<1, 1024, 0, stream>>>(straw, sumexp, disc, (float*)d_out);
}
